// Latent_38946763440842
// MI455X (gfx1250) — matrix operand order repaired and checked by reading, not compiled, not run on hardware
//
#include <hip/hip_runtime.h>
#include <hip/hip_bf16.h>

// Problem constants (from reference): B=16, D=256, H=32, W=32, M=2048
#define DQ   256
#define HWQ  1024          // H*W
#define NQ   16384         // B*H*W
#define MQ   2048

typedef __attribute__((ext_vector_type(16))) __bf16 v16bf;
typedef __attribute__((ext_vector_type(8)))  float  v8f;

__device__ __forceinline__ unsigned short f2bf(float f) {
    unsigned u = __float_as_uint(f);
    u += 0x7fffu + ((u >> 16) & 1u);     // round-to-nearest-even
    return (unsigned short)(u >> 16);
}

// Async DMA: 16 contiguous bytes global -> LDS, tracked by ASYNCcnt.
// VDST = LDS byte offset (low 32 bits of flat LDS pointer), VADDR = 64b global.
__device__ __forceinline__ void async_ld_b128(const float* gptr, void* lptr) {
    unsigned loff = (unsigned)(unsigned long long)lptr;
    asm volatile("global_load_async_to_lds_b128 %0, %1, off"
                 :: "v"(loff), "v"(gptr) : "memory");
}
__device__ __forceinline__ void wait_async0() {
    asm volatile("s_wait_asynccnt 0x0" ::: "memory");
}

// ---------------------------------------------------------------------------
// z_sq[n] = sum_d zf[n,d]^2   (zf[n,d] = z[b, d, h, w], d-stride = HW floats)
// ---------------------------------------------------------------------------
__global__ void zsq_kernel(const float* __restrict__ z, float* __restrict__ zsq) {
    int n = blockIdx.x * blockDim.x + threadIdx.x;     // 16384 threads
    int b = n >> 10, s = n & 1023;
    const float* p = z + (size_t)b * (DQ * HWQ) + s;
    float acc = 0.f;
    #pragma unroll 8
    for (int d = 0; d < DQ; ++d) { float v = p[(size_t)d * HWQ]; acc += v * v; }
    zsq[n] = acc;
}

// ---------------------------------------------------------------------------
// e_sq[m] = sum_d e[m,d]^2, one wave per row; also init per-column min to +inf
// ---------------------------------------------------------------------------
__global__ void esq_kernel(const float* __restrict__ e, float* __restrict__ esq,
                           unsigned* __restrict__ colmin) {
    int wave = (blockIdx.x * blockDim.x + threadIdx.x) >> 5;   // 2048 waves
    int lane = threadIdx.x & 31;
    const float* p = e + (size_t)wave * DQ;
    float acc = 0.f;
    #pragma unroll
    for (int d = lane; d < DQ; d += 32) { float v = p[d]; acc += v * v; }
    #pragma unroll
    for (int off = 16; off; off >>= 1) acc += __shfl_xor(acc, off, 32);
    if (lane == 0) { esq[wave] = acc; colmin[wave] = 0x7f800000u; }
}

// ---------------------------------------------------------------------------
// Fused: distance tile (128n x 128m) via bf16 WMMA + per-column min reduce.
// A tile: strided fp32 gather -> bf16 in LDS (register path).
// B tile: contiguous e rows DMA'd fp32 into LDS via global_load_async_to_lds.
// ---------------------------------------------------------------------------
__launch_bounds__(256)
__global__ void dist_kernel(const float* __restrict__ z, const float* __restrict__ e,
                            const float* __restrict__ zsq, const float* __restrict__ esq,
                            unsigned* __restrict__ colmin, float* __restrict__ out) {
    __shared__ __align__(16) unsigned short As[128][64];   // n x k tile (bf16 bits)
    __shared__ __align__(16) float          Bf[128][64];   // m x k tile (fp32, async-staged)

    const int tid  = threadIdx.x;
    const int lane = tid & 31;
    const int wav  = tid >> 5;
    const int wn   = wav >> 2;            // 0..1  (n-dir wave)
    const int wm   = wav & 3;             // 0..3  (m-dir wave)
    const int half = lane >> 4;           // 0/1
    const int l16  = lane & 15;

    const int n0 = blockIdx.y * 128;
    const int m0 = blockIdx.x * 128;

    v8f acc[4][2];
    #pragma unroll
    for (int i = 0; i < 4; ++i)
        #pragma unroll
        for (int j = 0; j < 2; ++j)
            acc[i][j] = (v8f)(0.f);

    // A staging assignment (coalesced over n, strided over d)
    const int a_n  = tid & 127;
    const int a_ds = tid >> 7;            // 0/1 -> d parity
    const int bA   = (n0 + a_n) >> 10;    // batch index
    const int sA   = (n0 + a_n) & 1023;   // spatial index
    const float* zb = z + (size_t)bA * (DQ * HWQ) + sA;

    // B async-staging assignment: 16 lanes x 4 floats cover one 64-wide k row
    const int b_d4 = (tid & 15) * 4;      // 0,4,...,60
    const int b_mb = tid >> 4;            // 0..15

    for (int k0 = 0; k0 < DQ; k0 += 64) {
        __syncthreads();                  // previous-iter fragment reads done
        // Kick off async DMA of the B tile (8 x b128 per thread)
        #pragma unroll
        for (int mm = b_mb; mm < 128; mm += 16)
            async_ld_b128(e + (size_t)(m0 + mm) * DQ + (k0 + b_d4), &Bf[mm][b_d4]);
        if (k0 + 64 < DQ)                 // prefetch next strided A chunk
            __builtin_prefetch(zb + (size_t)(k0 + 64) * HWQ, 0, 1);
        // Register-path A staging overlaps with the async B DMA
        #pragma unroll 8
        for (int dd = a_ds; dd < 64; dd += 2)
            As[a_n][dd] = f2bf(zb[(size_t)(k0 + dd) * HWQ]);
        wait_async0();                    // own async transfers landed in LDS
        __syncthreads();                  // all waves' tiles visible

        #pragma unroll
        for (int kc = 0; kc < 2; ++kc) {
            union { v16bf v; uint4 q[2]; } a[4];
            v16bf b[2];
            // A frag 16x32: lane(row=l16, half): k runs [half*8,+8) and [16+half*8,+8)
            #pragma unroll
            for (int i = 0; i < 4; ++i) {
                const unsigned short* row = &As[wn * 64 + i * 16 + l16][kc * 32];
                a[i].q[0] = *(const uint4*)(row + half * 8);
                a[i].q[1] = *(const uint4*)(row + 16 + half * 8);
            }
            // B frag 32x16: lane(col=l16, half): k runs [half*8,+8) and [16+half*8,+8), fp32 -> bf16
            #pragma unroll
            for (int j = 0; j < 2; ++j) {
                const float* row = &Bf[wm * 32 + j * 16 + l16][kc * 32 + half * 8];
                float4 f0 = *(const float4*)(row);
                float4 f1 = *(const float4*)(row + 4);
                float4 f2 = *(const float4*)(row + 16);
                float4 f3 = *(const float4*)(row + 20);
                v16bf bv;
                bv[0]  = (__bf16)f0.x; bv[1]  = (__bf16)f0.y;
                bv[2]  = (__bf16)f0.z; bv[3]  = (__bf16)f0.w;
                bv[4]  = (__bf16)f1.x; bv[5]  = (__bf16)f1.y;
                bv[6]  = (__bf16)f1.z; bv[7]  = (__bf16)f1.w;
                bv[8]  = (__bf16)f2.x; bv[9]  = (__bf16)f2.y;
                bv[10] = (__bf16)f2.z; bv[11] = (__bf16)f2.w;
                bv[12] = (__bf16)f3.x; bv[13] = (__bf16)f3.y;
                bv[14] = (__bf16)f3.z; bv[15] = (__bf16)f3.w;
                b[j] = bv;
            }
            #pragma unroll
            for (int i = 0; i < 4; ++i)
                #pragma unroll
                for (int j = 0; j < 2; ++j)
                    acc[i][j] = __builtin_amdgcn_wmma_f32_16x16x32_bf16(
                        false, a[i].v, false, b[j], (short)0, acc[i][j], false, false);
        }
    }

    // Epilogue: dist = max(zsq + esq - 2*cross, 0); stream out; fused column min
    #pragma unroll
    for (int j = 0; j < 2; ++j) {
        const int m = m0 + wm * 32 + j * 16 + l16;
        const float es = esq[m];
        float cmin = 3.402823466e+38f;
        #pragma unroll
        for (int i = 0; i < 4; ++i) {
            const int nb = n0 + wn * 64 + i * 16 + half * 8;   // C/D layout: half -> rows 8..15
            #pragma unroll
            for (int r = 0; r < 8; ++r) {
                const int n = nb + r;
                float d = fmaxf(zsq[n] + es - 2.0f * acc[i][j][r], 0.0f);
                out[(size_t)n * MQ + m] = d;
                cmin = fminf(cmin, d);
            }
        }
        // column j is split between lane l16 and l16+16 -> combine
        cmin = fminf(cmin, __shfl_xor(cmin, 16, 32));
        if (half == 0)
            atomicMin(&colmin[m], __float_as_uint(cmin));   // dist >= 0: uint order == float order
    }
}

// ---------------------------------------------------------------------------
// loss = mean over M of column mins
// ---------------------------------------------------------------------------
__global__ void loss_kernel(const unsigned* __restrict__ colmin, float* __restrict__ out) {
    __shared__ float red[256];
    float s = 0.f;
    for (int m = threadIdx.x; m < MQ; m += 256) s += __uint_as_float(colmin[m]);
    red[threadIdx.x] = s;
    __syncthreads();
    for (int off = 128; off; off >>= 1) {
        if (threadIdx.x < off) red[threadIdx.x] += red[threadIdx.x + off];
        __syncthreads();
    }
    if (threadIdx.x == 0) out[(size_t)NQ * MQ] = red[0] / (float)MQ;
}

// ---------------------------------------------------------------------------
extern "C" void kernel_launch(void* const* d_in, const int* in_sizes, int n_in,
                              void* d_out, int out_size, void* d_ws, size_t ws_size,
                              hipStream_t stream) {
    (void)in_sizes; (void)n_in; (void)out_size; (void)ws_size;
    const float* z = (const float*)d_in[0];   // (16, 256, 32, 32) fp32
    const float* e = (const float*)d_in[1];   // (2048, 256)       fp32
    float* out = (float*)d_out;               // dist (N*M) + loss (1)

    float*    zsq    = (float*)d_ws;          // 16384 floats
    float*    esq    = zsq + NQ;              // 2048  floats
    unsigned* cmin   = (unsigned*)(esq + MQ); // 2048  uints

    zsq_kernel<<<NQ / 256, 256, 0, stream>>>(z, zsq);
    esq_kernel<<<(MQ * 32) / 256, 256, 0, stream>>>(e, esq, cmin);
    dist_kernel<<<dim3(MQ / 128, NQ / 128), 256, 0, stream>>>(z, e, zsq, esq, cmin, out);
    loss_kernel<<<1, 256, 0, stream>>>(cmin, out);
}
